// CoreAttention_50586124812751
// MI455X (gfx1250) — matrix operand order repaired and checked by reading, not compiled, not run on hardware
//
#include <hip/hip_runtime.h>

// NTK hybrid causal attention for MI455X (gfx1250, wave32, WMMA bf16).
// SEQ=2048, BS=2, H=16, D=64.
// Pre-pass: K -> bf16 [bh][s][d], V -> bf16 transposed [bh][d][s] in d_ws.
// Main: flash-style causal pass + max-only tail pass (reference takes the
// row max over the FULL row before masking).

typedef __attribute__((ext_vector_type(16))) __bf16 v16bf;
typedef __attribute__((ext_vector_type(8)))  float  v8f;

#define SEQ    2048
#define BSZ    2
#define NH     16
#define DH     64
#define WAVES  4
#define KV_BLK 64
#define NBLK   (SEQ / KV_BLK)            // 32
#define ROWSTR (BSZ * NH * DH)           // 2048 floats between seq rows
#define KB_ELEMS ((size_t)BSZ * NH * SEQ * DH)   // 4 Mi ushorts = 8 MB

// float -> bf16 bits, round-to-nearest-even
__device__ __forceinline__ unsigned short f2bf(float f) {
    unsigned int u = __float_as_uint(f);
    return (unsigned short)((u + 0x7FFFu + ((u >> 16) & 1u)) >> 16);
}
// cheap round-half-up variant for attention weights
__device__ __forceinline__ unsigned short f2bf_fast(float f) {
    return (unsigned short)((__float_as_uint(f) + 0x8000u) >> 16);
}

union BfFrag {
    v16bf v;
    unsigned short u[16];
};

// ---------------- pre-pass: bf16 convert K, transpose+convert V ----------------
__global__ __launch_bounds__(256)
void prepass_kernel(const float* __restrict__ k_g, const float* __restrict__ v_g,
                    unsigned short* __restrict__ kb_ws,
                    unsigned short* __restrict__ vt_ws)
{
    __shared__ unsigned short tile[64][65];   // padded to dodge bank conflicts
    const int tid = threadIdx.x;
    const int sb  = blockIdx.x;   // 64-row seq block
    const int bh  = blockIdx.y;

    for (int i = tid; i < 64 * 64; i += 256) {
        int sl = i >> 6, d = i & 63;
        int s = sb * 64 + sl;
        size_t in = ((size_t)s * (BSZ * NH) + bh) * DH + d;
        kb_ws[((size_t)bh * SEQ + s) * DH + d] = f2bf(k_g[in]);
        tile[sl][d] = f2bf(v_g[in]);
    }
    __syncthreads();
    for (int i = tid; i < 64 * 64; i += 256) {
        int d = i >> 6, sl = i & 63;
        vt_ws[((size_t)bh * DH + d) * SEQ + sb * 64 + sl] = tile[sl][d];
    }
}

// S = Q x K^T over one 64-key block: four 16x16 tiles, K-dim 64 in two slices
__device__ __forceinline__ void compute_s(const unsigned short* k_lds,
                                          const BfFrag* qa, int g, int n,
                                          v8f s_t[4])
{
#pragma unroll
    for (int nt = 0; nt < 4; ++nt) {
        v8f acc;
#pragma unroll
        for (int r = 0; r < 8; ++r) acc[r] = 0.0f;
#pragma unroll
        for (int s = 0; s < 2; ++s) {
            BfFrag bf;   // B frag: Kc = (e>>3)*16+g*8+(e&7), col = nt*16+n
            const unsigned short* kp = &k_lds[(nt * 16 + n) * DH + s * 32];
#pragma unroll
            for (int e = 0; e < 16; ++e) bf.u[e] = kp[((e >> 3) * 16) + g * 8 + (e & 7)];
            acc = __builtin_amdgcn_wmma_f32_16x16x32_bf16(
                false, qa[s].v, false, bf.v, (short)0, acc, false, false);
        }
        s_t[nt] = acc;
    }
}

// ---------------- main attention kernel ----------------
__global__ __launch_bounds__(WAVES * 32)
void ntk_attn_kernel(const float* __restrict__ q_g,
                     const unsigned short* __restrict__ kb_ws,
                     const unsigned short* __restrict__ vt_ws,
                     const float* __restrict__ phik_g,
                     const float* __restrict__ phikv_g,
                     float* __restrict__ out_g)
{
    __shared__ __attribute__((aligned(16))) unsigned short k_lds[KV_BLK * DH];  // [key][d]
    __shared__ __attribute__((aligned(16))) unsigned short vt_lds[DH * KV_BLK]; // [d][key]
    __shared__ __attribute__((aligned(16))) unsigned short p_lds[WAVES][16 * KV_BLK];

    const int tid  = threadIdx.x;
    const int wave = tid >> 5;
    const int lane = tid & 31;
    const int g    = lane >> 4;
    const int n    = lane & 15;

    const int bh = blockIdx.y;
    const int b  = bh / NH;
    const int h  = bh % NH;
    const int bx = blockIdx.x;                 // workgroup query rows [bx*64, bx*64+64)
    const int q0 = (bx * WAVES + wave) * 16;
    const int bh_off = (b * NH + h) * DH;

    // ---- Q tile as two bf16 A-fragments, 1/sqrt(d) folded in ----
    BfFrag qa[2];
    {
        const float* qrow = q_g + (size_t)(q0 + n) * ROWSTR + bh_off;
#pragma unroll
        for (int s = 0; s < 2; ++s)
#pragma unroll
            for (int e = 0; e < 16; ++e) {
                int d = s * 32 + ((e >> 3) * 16) + g * 8 + (e & 7);
                qa[s].u[e] = f2bf(qrow[d] * 0.125f);
            }
    }

    // ones B-fragment for WMMA row-sums
    BfFrag ones;
#pragma unroll
    for (int e = 0; e < 16; ++e) ones.u[e] = 0x3F80;

    v8f o[4];
    float m[8], l[8];
#pragma unroll
    for (int t = 0; t < 4; ++t)
#pragma unroll
        for (int r = 0; r < 8; ++r) o[t][r] = 0.0f;
#pragma unroll
    for (int r = 0; r < 8; ++r) { m[r] = -1e30f; l[r] = 0.0f; }

    const unsigned short* kb_base = kb_ws + (size_t)bh * SEQ * DH;
    const unsigned short* vt_base = vt_ws + (size_t)bh * DH * SEQ;

    // =========== phase 1: causal blocks kb = 0 .. bx (uniform per WG) ===========
    for (int kb = 0; kb <= bx; ++kb) {
        __syncthreads();
        {   // stage K [key][d] and V^T [d][key] (bf16, pure copies)
            const uint4* ks = (const uint4*)(kb_base + (size_t)kb * KV_BLK * DH);
            uint4* kd = (uint4*)k_lds;
            const unsigned short* vs = vt_base + kb * KV_BLK;
            uint4* vd = (uint4*)vt_lds;
            for (int i = tid; i < (KV_BLK * DH) / 8; i += WAVES * 32) {
                kd[i] = ks[i];
                int d = i >> 3, c = i & 7;
                vd[i] = *(const uint4*)(vs + (size_t)d * SEQ + c * 8);
            }
        }
        if (kb < bx && tid < KV_BLK) {   // emits global_prefetch_b8
            __builtin_prefetch(kb_base + (size_t)(kb + 1) * KV_BLK * DH + tid * DH, 0, 3);
        }
        __syncthreads();

        v8f s_t[4];
        compute_s(k_lds, qa, g, n, s_t);

        // online max + rescale
        float mnew[8], fac[8];
#pragma unroll
        for (int r = 0; r < 8; ++r) {
            float loc = fmaxf(fmaxf(s_t[0][r], s_t[1][r]), fmaxf(s_t[2][r], s_t[3][r]));
            loc = fmaxf(loc, __shfl_xor(loc, 1));
            loc = fmaxf(loc, __shfl_xor(loc, 2));
            loc = fmaxf(loc, __shfl_xor(loc, 4));
            loc = fmaxf(loc, __shfl_xor(loc, 8));
            mnew[r] = fmaxf(m[r], loc);
            fac[r]  = __expf(m[r] - mnew[r]);
            m[r]    = mnew[r];
#pragma unroll
            for (int t = 0; t < 4; ++t) o[t][r] *= fac[r];
        }

        // P = exp(S - m), masked only on the diagonal block; stage to LDS (bf16)
        const bool diag = (kb == bx);
#pragma unroll
        for (int r = 0; r < 8; ++r) {
            int rr = r + 8 * g;
            int qrow = q0 + rr;
#pragma unroll
            for (int t = 0; t < 4; ++t) {
                float pv = __expf(s_t[t][r] - mnew[r]);
                if (diag) pv = (kb * KV_BLK + t * 16 + n <= qrow) ? pv : 0.0f;
                p_lds[wave][rr * KV_BLK + t * 16 + n] = f2bf_fast(pv);
            }
        }

        // P A-fragments (contiguous b128 reads)
        BfFrag pa[2];
#pragma unroll
        for (int sl = 0; sl < 2; ++sl)
#pragma unroll
            for (int e = 0; e < 16; ++e) {
                int kk = sl * 32 + ((e >> 3) * 16) + g * 8 + (e & 7);
                pa[sl].u[e] = p_lds[wave][n * KV_BLK + kk];
            }

        // row sums via WMMA against ones
        v8f rs;
#pragma unroll
        for (int r = 0; r < 8; ++r) rs[r] = 0.0f;
        rs = __builtin_amdgcn_wmma_f32_16x16x32_bf16(false, pa[0].v, false, ones.v,
                                                     (short)0, rs, false, false);
        rs = __builtin_amdgcn_wmma_f32_16x16x32_bf16(false, pa[1].v, false, ones.v,
                                                     (short)0, rs, false, false);
#pragma unroll
        for (int r = 0; r < 8; ++r) l[r] = l[r] * fac[r] + rs[r];

        // O += P x V  (V^T fragments contiguous)
#pragma unroll
        for (int t = 0; t < 4; ++t)
#pragma unroll
            for (int sl = 0; sl < 2; ++sl) {
                BfFrag vf;
                const unsigned short* vp = &vt_lds[(t * 16 + n) * KV_BLK + sl * 32];
#pragma unroll
                for (int e = 0; e < 16; ++e) vf.u[e] = vp[((e >> 3) * 16) + g * 8 + (e & 7)];
                o[t] = __builtin_amdgcn_wmma_f32_16x16x32_bf16(
                    false, pa[sl].v, false, vf.v, (short)0, o[t], false, false);
            }
    }

    // ====== phase 2: tail blocks (only the full-row max matters; defer reduce) ======
    float tmax[8];
#pragma unroll
    for (int r = 0; r < 8; ++r) tmax[r] = -1e30f;

    for (int kb = bx + 1; kb < NBLK; ++kb) {
        __syncthreads();
        {   // stage K only
            const uint4* ks = (const uint4*)(kb_base + (size_t)kb * KV_BLK * DH);
            uint4* kd = (uint4*)k_lds;
            for (int i = tid; i < (KV_BLK * DH) / 8; i += WAVES * 32) kd[i] = ks[i];
        }
        __syncthreads();

        v8f s_t[4];
        compute_s(k_lds, qa, g, n, s_t);
#pragma unroll
        for (int r = 0; r < 8; ++r)
            tmax[r] = fmaxf(tmax[r],
                            fmaxf(fmaxf(s_t[0][r], s_t[1][r]),
                                  fmaxf(s_t[2][r], s_t[3][r])));
    }

    // single cross-lane reduce + one final rescale
#pragma unroll
    for (int r = 0; r < 8; ++r) {
        float loc = tmax[r];
        loc = fmaxf(loc, __shfl_xor(loc, 1));
        loc = fmaxf(loc, __shfl_xor(loc, 2));
        loc = fmaxf(loc, __shfl_xor(loc, 4));
        loc = fmaxf(loc, __shfl_xor(loc, 8));
        float mf  = fmaxf(m[r], loc);
        float fac = __expf(m[r] - mf);
        m[r] = mf;
        l[r] *= fac;
#pragma unroll
        for (int t = 0; t < 4; ++t) o[t][r] *= fac;
    }

    // ==================== epilogue: phi(q) correction terms ====================
    BfFrag pq[2];
    float dotp = 0.0f;
    {
        const float* qrow = q_g + (size_t)(q0 + n) * ROWSTR + bh_off;
        const float* pk   = phik_g + h * DH;
#pragma unroll
        for (int s = 0; s < 2; ++s)
#pragma unroll
            for (int e = 0; e < 16; ++e) {
                int d   = s * 32 + ((e >> 3) * 16) + g * 8 + (e & 7);
                float x = qrow[d] * 0.35355339059327373f;   // q / d^0.25
                float ph = (x > 0.0f ? x : (__expf(x) - 1.0f)) + 1.0f;
                pq[s].u[e] = f2bf(ph);
                dotp += ph * fabsf(pk[d]);
            }
    }
    dotp += __shfl_xor(dotp, 16);   // lanes m and m+16 cover complementary d

    v8f ext[4];
#pragma unroll
    for (int t = 0; t < 4; ++t)
#pragma unroll
        for (int r = 0; r < 8; ++r) ext[t][r] = 0.0f;
#pragma unroll
    for (int t = 0; t < 4; ++t)
#pragma unroll
        for (int s = 0; s < 2; ++s) {
            BfFrag zf;   // B frag of Z: Kc(d) = s*32+(e>>3)*16+g*8+(e&7), col = t*16+n
            const float* zp = phikv_g + ((size_t)h * DH + s * 32) * DH + t * 16 + n;
#pragma unroll
            for (int e = 0; e < 16; ++e) zf.u[e] = f2bf(zp[(((e >> 3) * 16) + g * 8 + (e & 7)) * DH]);
            ext[t] = __builtin_amdgcn_wmma_f32_16x16x32_bf16(
                false, pq[s].v, false, zf.v, (short)0, ext[t], false, false);
        }

    // ctx = (O + ext*e^-m) / (l + dot*e^-m)
#pragma unroll
    for (int r = 0; r < 8; ++r) {
        int rr = r + 8 * g;
        float inv_e = __expf(-m[r]);
        float dot_r = __shfl(dotp, rr);
        float denom = l[r] + dot_r * inv_e;
        size_t base = (size_t)(q0 + rr) * ROWSTR + bh_off;
#pragma unroll
        for (int t = 0; t < 4; ++t)
            out_g[base + t * 16 + n] = (o[t][r] + ext[t][r] * inv_e) / denom;
    }
}

extern "C" void kernel_launch(void* const* d_in, const int* in_sizes, int n_in,
                              void* d_out, int out_size, void* d_ws, size_t ws_size,
                              hipStream_t stream) {
    const float* q     = (const float*)d_in[0];
    const float* k     = (const float*)d_in[1];
    const float* v     = (const float*)d_in[2];
    const float* phik  = (const float*)d_in[4];
    const float* phikv = (const float*)d_in[5];
    float* out = (float*)d_out;

    unsigned short* kb_ws = (unsigned short*)d_ws;          // 8 MB
    unsigned short* vt_ws = kb_ws + KB_ELEMS;               // 8 MB

    dim3 pgrid(SEQ / 64, BSZ * NH);
    prepass_kernel<<<pgrid, 256, 0, stream>>>(k, v, kb_ws, vt_ws);

    dim3 grid(SEQ / (16 * WAVES), BSZ * NH);   // (32, 32)
    ntk_attn_kernel<<<grid, dim3(WAVES * 32), 0, stream>>>(q, kb_ws, vt_ws,
                                                           phik, phikv, out);
}
